// MultiheadAttention_7267084664989
// MI455X (gfx1250) — matrix operand order repaired and checked by reading, not compiled, not run on hardware
//
#include <hip/hip_runtime.h>

// ---------------------------------------------------------------------------
// Problem constants (fixed by the reference)
// ---------------------------------------------------------------------------
constexpr int S_  = 2048;   // sequence length
constexpr int D_  = 2048;   // model dim
constexpr int H_  = 16;     // heads
constexpr int KVH_ = 4;     // kv heads
constexpr int HD_ = 128;    // head dim
constexpr int B_  = 2;      // batch
constexpr int M_TOT = B_ * S_;        // 4096 rows
constexpr int KV_FEAT = 2 * HD_ * KVH_; // 1024

typedef _Float16 v16h __attribute__((ext_vector_type(16)));
typedef _Float16 v8h  __attribute__((ext_vector_type(8)));
typedef float    v8f  __attribute__((ext_vector_type(8)));

// ---------------------------------------------------------------------------
// WMMA helpers (CDNA5: V_WMMA_F32_16X16X32_F16, wave32)
// ---------------------------------------------------------------------------
static __device__ __forceinline__ v8f wmma16(v16h a, v16h b, v8f c) {
  // (neg_a, A, neg_b, B, c_mod, C, reuse_a, reuse_b)
  return __builtin_amdgcn_wmma_f32_16x16x32_f16(false, a, false, b, (short)0, c,
                                                false, false);
}

// A-fragment, 16x32 f16, row-major source with leading dim lda (halfs).
// ISA layout: lane = 16*half + m ; elems 0..7 -> K = half*8+e ; 8..15 -> K = 16+half*8+e
static __device__ __forceinline__ v16h load_afrag(const _Float16* __restrict__ A,
                                                  int lda) {
  int l = threadIdx.x & 31;
  int m = l & 15, hf = l >> 4;
  v8h lo = *(const v8h*)(A + (size_t)m * lda + hf * 8);
  v8h hi = *(const v8h*)(A + (size_t)m * lda + 16 + hf * 8);
  v16h r;
#pragma unroll
  for (int e = 0; e < 8; ++e) { r[e] = lo[e]; r[e + 8] = hi[e]; }
  return r;
}

// B-fragment, 32x16 f16, from B^T stored row-major (row n = output col, ld = ldb).
// ISA layout: lane = 16*kh + n ; elems 0..7 -> K = kh*8+e ; 8..15 -> K = 16+kh*8+e
static __device__ __forceinline__ v16h load_bfrag(const _Float16* __restrict__ BT,
                                                  int ldb) {
  int l = threadIdx.x & 31;
  int n = l & 15, kh = l >> 4;
  v8h lo = *(const v8h*)(BT + (size_t)n * ldb + kh * 8);
  v8h hi = *(const v8h*)(BT + (size_t)n * ldb + 16 + kh * 8);
  v16h r;
#pragma unroll
  for (int e = 0; e < 8; ++e) { r[e] = lo[e]; r[e + 8] = hi[e]; }
  return r;
}

// butterfly reductions within each 16-lane half of a wave32
static __device__ __forceinline__ float redmax16(float v) {
#pragma unroll
  for (int m = 8; m >= 1; m >>= 1) v = fmaxf(v, __shfl_xor(v, m, 32));
  return v;
}
static __device__ __forceinline__ float redsum16(float v) {
#pragma unroll
  for (int m = 8; m >= 1; m >>= 1) v += __shfl_xor(v, m, 32);
  return v;
}

// ---------------------------------------------------------------------------
// Kernel 1: per-token gating  cw = sigmoid(sum |x|), emit f16 gated + plain x
// ---------------------------------------------------------------------------
__global__ __launch_bounds__(256) void gate_convert(const float* __restrict__ x,
                                                    _Float16* __restrict__ xg16,
                                                    _Float16* __restrict__ xn16) {
  __shared__ float red[256];
  const int row = blockIdx.x;
  const float* xr = x + (size_t)row * D_;
  float s = 0.f;
  for (int c = threadIdx.x; c < D_; c += 256) s += fabsf(xr[c]);
  red[threadIdx.x] = s;
  __syncthreads();
#pragma unroll
  for (int st = 128; st > 0; st >>= 1) {
    if (threadIdx.x < st) red[threadIdx.x] += red[threadIdx.x + st];
    __syncthreads();
  }
  const float cw = 1.f / (1.f + __expf(-red[0]));
  for (int c = threadIdx.x; c < D_; c += 256) {
    float v = xr[c];
    xg16[(size_t)row * D_ + c] = (_Float16)(v * cw);
    xn16[(size_t)row * D_ + c] = (_Float16)v;
  }
}

// ---------------------------------------------------------------------------
// Kernel 2: W[k][n] (f32) -> W^T[n][k] (f16)
// ---------------------------------------------------------------------------
__global__ __launch_bounds__(256) void transpose_f16(const float* __restrict__ W,
                                                     _Float16* __restrict__ WT,
                                                     int K, int N) {
  size_t idx = (size_t)blockIdx.x * 256 + threadIdx.x;
  if (idx >= (size_t)K * N) return;
  int k = (int)(idx / N), n = (int)(idx % N);
  WT[(size_t)n * K + k] = (_Float16)W[idx];
}

// ---------------------------------------------------------------------------
// Kernel 3: WMMA GEMM  C[M,N] = A[M,K] (f16, row-major) * B (B^T f16 row-major)
// Block tile 64x64, 4 waves, each wave: 16 rows x 64 cols, K-step 32.
// mode 0: f32 out + bias          (final projection)
// mode 1: f16 out, q head layout  [B][H][S][HD]
// mode 2: f16 out, split kv: K -> [B][KVH][S][HD], V -> [B][KVH][HD][S]
// ---------------------------------------------------------------------------
__global__ __launch_bounds__(128) void gemm_wmma(
    const _Float16* __restrict__ A, const _Float16* __restrict__ BT,
    int M, int N, int K, int mode, float* __restrict__ Cf,
    const float* __restrict__ bias, _Float16* __restrict__ O16a,
    _Float16* __restrict__ O16b) {
  const int wv = threadIdx.x >> 5;
  const int l = threadIdx.x & 31;
  const int m0 = blockIdx.y * 64 + wv * 16;
  const int n0 = blockIdx.x * 64;

  v8f acc[4];
#pragma unroll
  for (int t = 0; t < 4; ++t)
#pragma unroll
    for (int r = 0; r < 8; ++r) acc[t][r] = 0.f;

  const _Float16* Ab = A + (size_t)m0 * K;
  for (int k0 = 0; k0 < K; k0 += 32) {
    v16h a = load_afrag(Ab + k0, K);
#pragma unroll
    for (int t = 0; t < 4; ++t) {
      v16h b = load_bfrag(BT + (size_t)(n0 + t * 16) * K + k0, K);
      acc[t] = wmma16(a, b, acc[t]);
    }
  }

  const int nl = l & 15, mh = l >> 4;
  if (mode == 0) {
#pragma unroll
    for (int t = 0; t < 4; ++t) {
      int col = n0 + t * 16 + nl;
      float bv = bias[col];
#pragma unroll
      for (int r = 0; r < 8; ++r) {
        int row = m0 + mh * 8 + r;
        Cf[(size_t)row * N + col] = acc[t][r] + bv;
      }
    }
  } else if (mode == 1) {
#pragma unroll
    for (int t = 0; t < 4; ++t) {
      int c = n0 + t * 16 + nl;
      int h = c >> 7, d = c & 127;
#pragma unroll
      for (int r = 0; r < 8; ++r) {
        int row = m0 + mh * 8 + r;
        int bb = row >> 11, s = row & (S_ - 1);
        O16a[(((size_t)(bb * H_ + h) * S_) + s) * HD_ + d] = (_Float16)acc[t][r];
      }
    }
  } else {  // mode 2: kv split
#pragma unroll
    for (int t = 0; t < 4; ++t) {
      int c = n0 + t * 16 + nl;
#pragma unroll
      for (int r = 0; r < 8; ++r) {
        int row = m0 + mh * 8 + r;
        int bb = row >> 11, s = row & (S_ - 1);
        if (c < KVH_ * HD_) {
          int kvh = c >> 7, d = c & 127;
          O16a[(((size_t)(bb * KVH_ + kvh) * S_) + s) * HD_ + d] =
              (_Float16)acc[t][r];
        } else {
          int c2 = c - KVH_ * HD_;
          int kvh = c2 >> 7, d = c2 & 127;
          O16b[(((size_t)(bb * KVH_ + kvh) * HD_) + d) * S_ + s] =
              (_Float16)acc[t][r];
        }
      }
    }
  }
}

// ---------------------------------------------------------------------------
// Kernel 4: flash attention with inverted window mask + ALiBi.
// Block = 128 threads (4 waves); wave handles 16 query rows, full hd=128.
// Key loop: 32 keys/step -> 8 QK WMMAs + 8 PV WMMAs.
// ---------------------------------------------------------------------------
__global__ __launch_bounds__(128) void attn_wmma(
    const _Float16* __restrict__ q16, const _Float16* __restrict__ k16,
    const _Float16* __restrict__ v16t, _Float16* __restrict__ ao16) {
  __shared__ _Float16 lds_p[4 * 16 * 32];  // per-wave 16x32 P tile

  const int wv = threadIdx.x >> 5;
  const int l = threadIdx.x & 31;
  const int rb = blockIdx.x & 31;   // S/64 = 32 row-blocks
  const int bh = blockIdx.x >> 5;   // b*H + h
  const int b = bh >> 4, h = bh & 15;
  const int i0 = rb * 64 + wv * 16;
  const int kvh = h >> 2;  // GQA repeat r = 4
  const float slope = exp2f(-0.5f * (float)(h + 1));  // base 2^-0.5, H=16
  const float scale = 0.08838834764831845f;           // 1/sqrt(128)

  const _Float16* qb = q16 + ((size_t)bh * S_ + i0) * HD_;
  const _Float16* kb = k16 + ((size_t)(b * KVH_ + kvh) * S_) * HD_;
  const _Float16* vb = v16t + ((size_t)(b * KVH_ + kvh) * HD_) * S_;

  v16h qa[4];
#pragma unroll
  for (int dc = 0; dc < 4; ++dc) qa[dc] = load_afrag(qb + dc * 32, HD_);

  v8f acc[8];
#pragma unroll
  for (int t = 0; t < 8; ++t)
#pragma unroll
    for (int r = 0; r < 8; ++r) acc[t][r] = 0.f;

  float row_m[8], row_l[8];
#pragma unroll
  for (int r = 0; r < 8; ++r) { row_m[r] = -3.0e38f; row_l[r] = 0.f; }

  const int nl = l & 15, mh = l >> 4;
  _Float16* lp = lds_p + wv * (16 * 32);

  for (int jt = 0; jt < S_; jt += 32) {
    v8f s0, s1;
#pragma unroll
    for (int r = 0; r < 8; ++r) { s0[r] = 0.f; s1[r] = 0.f; }
#pragma unroll
    for (int dc = 0; dc < 4; ++dc) {
      v16h bk0 = load_bfrag(kb + (size_t)jt * HD_ + dc * 32, HD_);
      v16h bk1 = load_bfrag(kb + (size_t)(jt + 16) * HD_ + dc * 32, HD_);
      s0 = wmma16(qa[dc], bk0, s0);
      s1 = wmma16(qa[dc], bk1, s1);
    }

    float alpha[8];
#pragma unroll
    for (int r = 0; r < 8; ++r) {
      int i = i0 + mh * 8 + r;
      int j0 = jt + nl, j1 = jt + 16 + nl;
      int d0 = i > j0 ? i - j0 : j0 - i;
      int d1 = i > j1 ? i - j1 : j1 - i;
      // reference: allowed (window|global) positions are REPLACED by -1e9
      bool a0 = (d0 <= 256) || (i < 64) || (j0 < 64);
      bool a1 = (d1 <= 256) || (i < 64) || (j1 < 64);
      float v0 = a0 ? -1.0e9f : (s0[r] * scale - slope * (float)d0);
      float v1 = a1 ? -1.0e9f : (s1[r] * scale - slope * (float)d1);
      float tm = redmax16(fmaxf(v0, v1));
      float mn = fmaxf(row_m[r], tm);
      float al = __expf(row_m[r] - mn);
      float p0 = __expf(v0 - mn);
      float p1 = __expf(v1 - mn);
      row_l[r] = row_l[r] * al + redsum16(p0 + p1);
      row_m[r] = mn;
      alpha[r] = al;
      // stage P (D-frag layout -> LDS, re-read below in A-frag layout)
      lp[(mh * 8 + r) * 32 + nl] = (_Float16)p0;
      lp[(mh * 8 + r) * 32 + 16 + nl] = (_Float16)p1;
    }
#pragma unroll
    for (int t = 0; t < 8; ++t)
#pragma unroll
      for (int r = 0; r < 8; ++r) acc[t][r] *= alpha[r];

    // drain LDS stores before cross-lane re-read (same-wave DS ops are in-order)
    asm volatile("s_wait_dscnt 0x0" ::: "memory");

    v16h pa;
#pragma unroll
    for (int e = 0; e < 8; ++e) {
      pa[e] = lp[nl * 32 + mh * 8 + e];            // m = nl, half = mh
      pa[e + 8] = lp[nl * 32 + 16 + mh * 8 + e];
    }

#pragma unroll
    for (int t = 0; t < 8; ++t) {
      v16h bv = load_bfrag(vb + (size_t)(t * 16) * S_ + jt, S_);
      acc[t] = wmma16(pa, bv, acc[t]);
    }
  }

  float inv[8];
#pragma unroll
  for (int r = 0; r < 8; ++r) inv[r] = 1.f / row_l[r];
#pragma unroll
  for (int t = 0; t < 8; ++t)
#pragma unroll
    for (int r = 0; r < 8; ++r) {
      int i = i0 + mh * 8 + r;
      ao16[((size_t)b * S_ + i) * D_ + h * HD_ + t * 16 + nl] =
          (_Float16)(acc[t][r] * inv[r]);
    }
}

// ---------------------------------------------------------------------------
// Host launch
// ---------------------------------------------------------------------------
extern "C" void kernel_launch(void* const* d_in, const int* in_sizes, int n_in,
                              void* d_out, int out_size, void* d_ws,
                              size_t ws_size, hipStream_t stream) {
  const float* x = (const float*)d_in[0];
  const float* Wq = (const float*)d_in[1];
  const float* Wkv = (const float*)d_in[2];
  const float* Wo = (const float*)d_in[3];
  const float* bo = (const float*)d_in[4];

  char* w = (char*)d_ws;
  // workspace layout (bytes, 256-aligned)
  _Float16* xg16 = (_Float16*)(w + 0);          // 16,777,216
  _Float16* xn16 = (_Float16*)(w + 16777216);   // 16,777,216
  _Float16* WqT  = (_Float16*)(w + 33554432);   //  8,388,608
  _Float16* WkvT = (_Float16*)(w + 41943040);   //  4,194,304
  _Float16* WoT  = (_Float16*)(w + 46137344);   //  8,388,608
  _Float16* q16  = (_Float16*)(w + 54525952);   // 16,777,216
  _Float16* k16  = (_Float16*)(w + 71303168);   //  4,194,304
  _Float16* v16t = (_Float16*)(w + 75497472);   //  4,194,304
  _Float16* ao16 = (_Float16*)(w + 79691776);   // 16,777,216  (total ~96.5MB)

  // weights -> f16 transposed
  transpose_f16<<<(D_ * D_ + 255) / 256, 256, 0, stream>>>(Wq, WqT, D_, D_);
  transpose_f16<<<(D_ * KV_FEAT + 255) / 256, 256, 0, stream>>>(Wkv, WkvT, D_,
                                                                KV_FEAT);
  transpose_f16<<<(D_ * D_ + 255) / 256, 256, 0, stream>>>(Wo, WoT, D_, D_);

  // gating + f16 conversion of activations
  gate_convert<<<M_TOT, 256, 0, stream>>>(x, xg16, xn16);

  // q = (x*cw) @ Wq  -> [B][H][S][HD] f16
  gemm_wmma<<<dim3(D_ / 64, M_TOT / 64), 128, 0, stream>>>(
      xg16, WqT, M_TOT, D_, D_, 1, nullptr, nullptr, q16, nullptr);

  // kv = x @ Wkv -> K [B][KVH][S][HD], V^T [B][KVH][HD][S] f16
  gemm_wmma<<<dim3(KV_FEAT / 64, M_TOT / 64), 128, 0, stream>>>(
      xn16, WkvT, M_TOT, KV_FEAT, D_, 2, nullptr, nullptr, k16, v16t);

  // flash attention (inverted mask + ALiBi) -> [B][S][D] f16
  attn_wmma<<<B_ * H_ * (S_ / 64), 128, 0, stream>>>(q16, k16, v16t, ao16);

  // out = ao @ Wo + bo -> f32
  gemm_wmma<<<dim3(D_ / 64, M_TOT / 64), 128, 0, stream>>>(
      ao16, WoT, M_TOT, D_, D_, 0, (float*)d_out, bo, nullptr, nullptr);
}
